// SharableMultiheadAttention_3513283248898
// MI455X (gfx1250) — matrix operand order repaired and checked by reading, not compiled, not run on hardware
//
#include <hip/hip_runtime.h>
#include <hip/hip_bf16.h>

// ---------------------------------------------------------------------------
// Multi-head attention forward for MI455X (gfx1250, wave32, WMMA).
//   EMBED=1024, HEADS=16, DH=64, SEQ=2048, BATCH=2
// Outputs (concatenated in d_out): out (2048,2,1024) f32 ; attn (2,16,2048,2048) f32
// ---------------------------------------------------------------------------

typedef __attribute__((ext_vector_type(16))) _Float16 v16h;
typedef __attribute__((ext_vector_type(8)))  _Float16 v8h;
typedef __attribute__((ext_vector_type(8)))  float    v8f;
typedef __attribute__((ext_vector_type(4)))  int      v4i;

union Frag16 { v16h v; v8h h[2]; };

#define LDT 40              // padded LDS stride in halves for BK=32 tiles
#define LDK 72              // padded LDS stride in halves for K=64 tiles
#define SEQ 2048
#define EMBED 1024
#define NBH 32              // BATCH * HEADS
#define DH 64

// --- gfx1250 async global->LDS copy (guarded; plain copy fallback) ----------
#if __has_builtin(__builtin_amdgcn_global_load_async_to_lds_b128)
#define HAVE_ASYNC_COPY 1
#define ASYNC_CP_B128(gsrc, ldst)                                              \
    __builtin_amdgcn_global_load_async_to_lds_b128(                            \
        (__attribute__((address_space(1))) v4i*)(gsrc),                        \
        (__attribute__((address_space(3))) v4i*)(ldst), 0, 0)
#else
#define HAVE_ASYNC_COPY 0
#define ASYNC_CP_B128(gsrc, ldst)                                              \
    (*(uint4*)(ldst) = *(const uint4*)(gsrc))
#endif

__device__ __forceinline__ void wait_async_lds() {
#if HAVE_ASYNC_COPY
#if __has_builtin(__builtin_amdgcn_s_wait_asynccnt)
    __builtin_amdgcn_s_wait_asynccnt(0);
#else
    asm volatile("s_wait_asynccnt 0x0" ::: "memory");
#endif
#endif
}

// --- fragment loads from LDS (wave32 layouts per CDNA5 ISA 7.12.2) ----------
// A (16x32 f16): lane l -> row m=l%16, hg=l/16; halves 0..7 = K[hg*8 .. +7],
//                halves 8..15 = K[16+hg*8 .. +7].
__device__ __forceinline__ v16h frag_a(const _Float16* base, int ld, int m,
                                       int k0, int hg) {
    Frag16 f;
    const _Float16* p = base + m * ld + k0;
    f.h[0] = *(const v8h*)(p + hg * 8);
    f.h[1] = *(const v8h*)(p + 16 + hg * 8);
    return f.v;
}
// B (32x16 f16), LDS stored [N][K]: lane l -> col n=l%16, hg=l/16;
//                halves 0..7 = K[hg*8 .. +7], halves 8..15 = K[16+hg*8 .. +7].
__device__ __forceinline__ v16h frag_b(const _Float16* base, int ld, int n,
                                       int k0, int hg) {
    Frag16 f;
    const _Float16* p = base + n * ld + k0;
    f.h[0] = *(const v8h*)(p + hg * 8);
    f.h[1] = *(const v8h*)(p + 16 + hg * 8);
    return f.v;
}

#define WMMA_F16(a, b, c) \
    __builtin_amdgcn_wmma_f32_16x16x32_f16(false, (a), false, (b), (short)0, (c), false, false)

// ---------------------------------------------------------------------------
// 1) Projection: Y[bh, n, dh] (f16) = X(nb,1024) @ W(e,1024)^T  * scale
//    grid (32, 16), 256 threads.
// ---------------------------------------------------------------------------
__global__ __launch_bounds__(256) void proj_kernel(const float* __restrict__ X,
                                                   const float* __restrict__ W,
                                                   _Float16* __restrict__ Y,
                                                   float scale) {
    __shared__ _Float16 As[128 * LDT];
    __shared__ _Float16 Bs[64 * LDT];

    const int tid  = threadIdx.x;
    const int lane = tid & 31;
    const int wv   = tid >> 5;
    const int wm   = (wv & 3) * 32;
    const int wn   = (wv >> 2) * 32;
    const int lm   = lane & 15;
    const int hg   = lane >> 4;
    const int m0   = blockIdx.x * 128;
    const int n0   = blockIdx.y * 64;

    const int r = tid >> 3;          // 0..31
    const int c = (tid & 7) * 4;     // 0..28 step 4

    v8f acc[2][2] = {};

    for (int k0 = 0; k0 < EMBED; k0 += 32) {
        if (k0 + 32 < EMBED) {       // prefetch next K-tile into near caches
            __builtin_prefetch(X + (size_t)(m0 + r) * EMBED + k0 + 32 + c, 0, 3);
            __builtin_prefetch(W + (size_t)(n0 + r) * EMBED + k0 + 32 + c, 0, 3);
        }
#pragma unroll
        for (int p = 0; p < 4; ++p) {    // A tile: 128x32, f32 -> f16
            const int row = p * 32 + r;
            const float4 x = *(const float4*)(X + (size_t)(m0 + row) * EMBED + k0 + c);
            _Float16* d = As + row * LDT + c;
            d[0] = (_Float16)x.x; d[1] = (_Float16)x.y;
            d[2] = (_Float16)x.z; d[3] = (_Float16)x.w;
        }
#pragma unroll
        for (int p = 0; p < 2; ++p) {    // B tile: 64x32 from W rows (K-contig)
            const int row = p * 32 + r;
            const float4 x = *(const float4*)(W + (size_t)(n0 + row) * EMBED + k0 + c);
            _Float16* d = Bs + row * LDT + c;
            d[0] = (_Float16)x.x; d[1] = (_Float16)x.y;
            d[2] = (_Float16)x.z; d[3] = (_Float16)x.w;
        }
        __syncthreads();

        v16h a[2], b[2];
        a[0] = frag_a(As, LDT, wm + lm, 0, hg);
        a[1] = frag_a(As, LDT, wm + 16 + lm, 0, hg);
        b[0] = frag_b(Bs, LDT, wn + lm, 0, hg);
        b[1] = frag_b(Bs, LDT, wn + 16 + lm, 0, hg);
#pragma unroll
        for (int fm = 0; fm < 2; ++fm)
#pragma unroll
            for (int fn = 0; fn < 2; ++fn)
                acc[fm][fn] = WMMA_F16(a[fm], b[fn], acc[fm][fn]);
        __syncthreads();
    }

    // epilogue: scatter f16 into (bh, n, dh) layout
#pragma unroll
    for (int fm = 0; fm < 2; ++fm)
#pragma unroll
        for (int fn = 0; fn < 2; ++fn)
#pragma unroll
            for (int j = 0; j < 8; ++j) {
                const int gm = m0 + wm + fm * 16 + (j + 8 * hg);  // row nb
                const int gn = n0 + wn + fn * 16 + lm;            // col e
                const int nseq = gm >> 1, bb = gm & 1;
                const int hh = gn >> 6, dh = gn & 63;
                Y[((size_t)(bb * 16 + hh) * SEQ + nseq) * DH + dh] =
                    (_Float16)(acc[fm][fn][j] * scale);
            }
}

// ---------------------------------------------------------------------------
// 2) Scores: S[bh, i, j] (f32) = Qh[bh,i,:] . Kh[bh,j,:]   (K = 64, single tile)
//    grid (16, 32, 32), 256 threads. Full-K LDS tiles, async copies, 1 barrier.
// ---------------------------------------------------------------------------
__global__ __launch_bounds__(256) void scores_kernel(const _Float16* __restrict__ Qh,
                                                     const _Float16* __restrict__ Kh,
                                                     float* __restrict__ S) {
    __shared__ _Float16 As[128 * LDK];
    __shared__ _Float16 Bs[64 * LDK];

    const int tid  = threadIdx.x;
    const int lane = tid & 31;
    const int wv   = tid >> 5;
    const int wm   = (wv & 3) * 32;
    const int wn   = (wv >> 2) * 32;
    const int lm   = lane & 15;
    const int hg   = lane >> 4;
    const int m0   = blockIdx.x * 128;
    const int n0   = blockIdx.y * 64;
    const int bh   = blockIdx.z;

    const _Float16* Qb = Qh + (size_t)bh * SEQ * DH;
    const _Float16* Kb = Kh + (size_t)bh * SEQ * DH;
    float* Sb = S + (size_t)bh * SEQ * SEQ;

    {   // A: 128 rows x 64 halves (2 threads/row, 64B each = 4x b128)
        const int ar = tid >> 1;
        const int ac = (tid & 1) * 32;
        const _Float16* g = Qb + (size_t)(m0 + ar) * DH + ac;
        _Float16* l = As + ar * LDK + ac;
#pragma unroll
        for (int q = 0; q < 4; ++q) ASYNC_CP_B128(g + q * 8, l + q * 8);
        // B: 64 rows x 64 halves (4 threads/row, 32B each = 2x b128)
        const int br = tid >> 2;
        const int bc = (tid & 3) * 16;
        const _Float16* gb = Kb + (size_t)(n0 + br) * DH + bc;
        _Float16* lb = Bs + br * LDK + bc;
#pragma unroll
        for (int q = 0; q < 2; ++q) ASYNC_CP_B128(gb + q * 8, lb + q * 8);
        wait_async_lds();
    }
    __syncthreads();

    v8f acc[2][2] = {};
#pragma unroll
    for (int k0 = 0; k0 < DH; k0 += 32) {
        v16h a[2], b[2];
        a[0] = frag_a(As, LDK, wm + lm, k0, hg);
        a[1] = frag_a(As, LDK, wm + 16 + lm, k0, hg);
        b[0] = frag_b(Bs, LDK, wn + lm, k0, hg);
        b[1] = frag_b(Bs, LDK, wn + 16 + lm, k0, hg);
#pragma unroll
        for (int fm = 0; fm < 2; ++fm)
#pragma unroll
            for (int fn = 0; fn < 2; ++fn)
                acc[fm][fn] = WMMA_F16(a[fm], b[fn], acc[fm][fn]);
    }

#pragma unroll
    for (int fm = 0; fm < 2; ++fm)
#pragma unroll
        for (int fn = 0; fn < 2; ++fn)
#pragma unroll
            for (int j = 0; j < 8; ++j) {
                const int gi = m0 + wm + fm * 16 + (j + 8 * hg);
                const int gj = n0 + wn + fn * 16 + lm;
                Sb[(size_t)gi * SEQ + gj] = acc[fm][fn][j];
            }
}

// ---------------------------------------------------------------------------
// 3) Row softmax in place. grid = 65536 rows, 256 threads/row, 8 elems/thread.
// ---------------------------------------------------------------------------
__global__ __launch_bounds__(256) void softmax_kernel(float* __restrict__ S) {
    __shared__ float red[256];
    float* row = S + (size_t)blockIdx.x * SEQ;
    const int tid = threadIdx.x;

    float v[8];
    float mx = -3.0e38f;
#pragma unroll
    for (int t = 0; t < 8; ++t) {
        v[t] = row[tid + t * 256];
        mx = fmaxf(mx, v[t]);
    }
    red[tid] = mx;
    __syncthreads();
    for (int s = 128; s > 0; s >>= 1) {
        if (tid < s) red[tid] = fmaxf(red[tid], red[tid + s]);
        __syncthreads();
    }
    mx = red[0];
    __syncthreads();

    float sum = 0.0f;
#pragma unroll
    for (int t = 0; t < 8; ++t) {
        v[t] = __expf(v[t] - mx);
        sum += v[t];
    }
    red[tid] = sum;
    __syncthreads();
    for (int s = 128; s > 0; s >>= 1) {
        if (tid < s) red[tid] += red[tid + s];
        __syncthreads();
    }
    const float inv = 1.0f / red[0];
#pragma unroll
    for (int t = 0; t < 8; ++t) row[tid + t * 256] = v[t] * inv;
}

// ---------------------------------------------------------------------------
// 4) Context: Ctx[nb, e] (f16) = attn[bh] (2048x2048 f32) @ Vh[bh] (2048x64)
//    grid (16, 32), 256 threads. BN = 64 covers full head dim.
// ---------------------------------------------------------------------------
__global__ __launch_bounds__(256) void attnv_kernel(const float* __restrict__ S,
                                                    const _Float16* __restrict__ Vh,
                                                    _Float16* __restrict__ Ctx) {
    __shared__ _Float16 As[128 * LDT];
    __shared__ _Float16 Bs[64 * LDT];

    const int tid  = threadIdx.x;
    const int lane = tid & 31;
    const int wv   = tid >> 5;
    const int wm   = (wv & 3) * 32;
    const int wn   = (wv >> 2) * 32;
    const int lm   = lane & 15;
    const int hg   = lane >> 4;
    const int m0   = blockIdx.x * 128;
    const int bh   = blockIdx.y;

    const float* Sb = S + (size_t)bh * SEQ * SEQ;
    const _Float16* Vb = Vh + (size_t)bh * SEQ * DH;

    const int r   = tid >> 3;
    const int c   = (tid & 7) * 4;
    const int jl  = tid >> 3;          // V-tile row (k)
    const int dh0 = (tid & 7) * 8;     // V-tile cols

    v8f acc[2][2] = {};

    for (int k0 = 0; k0 < SEQ; k0 += 32) {
        if (k0 + 32 < SEQ)             // prefetch next attn K-tile into near caches
            __builtin_prefetch(Sb + (size_t)(m0 + r) * SEQ + k0 + 32 + c, 0, 3);
#pragma unroll
        for (int p = 0; p < 4; ++p) {    // A: attn f32 -> f16
            const int row = p * 32 + r;
            const float4 x = *(const float4*)(Sb + (size_t)(m0 + row) * SEQ + k0 + c);
            _Float16* d = As + row * LDT + c;
            d[0] = (_Float16)x.x; d[1] = (_Float16)x.y;
            d[2] = (_Float16)x.z; d[3] = (_Float16)x.w;
        }
        {   // B: transpose V tile (32 x 64) -> Bs[dh][k]
            const v8h val = *(const v8h*)(Vb + (size_t)(k0 + jl) * DH + dh0);
#pragma unroll
            for (int q = 0; q < 8; ++q) Bs[(dh0 + q) * LDT + jl] = val[q];
        }
        __syncthreads();

        v16h a[2], b[2];
        a[0] = frag_a(As, LDT, wm + lm, 0, hg);
        a[1] = frag_a(As, LDT, wm + 16 + lm, 0, hg);
        b[0] = frag_b(Bs, LDT, wn + lm, 0, hg);
        b[1] = frag_b(Bs, LDT, wn + 16 + lm, 0, hg);
#pragma unroll
        for (int fm = 0; fm < 2; ++fm)
#pragma unroll
            for (int fn = 0; fn < 2; ++fn)
                acc[fm][fn] = WMMA_F16(a[fm], b[fn], acc[fm][fn]);
        __syncthreads();
    }

    const int bb = bh >> 4, hh = bh & 15;
#pragma unroll
    for (int fm = 0; fm < 2; ++fm)
#pragma unroll
        for (int fn = 0; fn < 2; ++fn)
#pragma unroll
            for (int j = 0; j < 8; ++j) {
                const int gi = m0 + wm + fm * 16 + (j + 8 * hg);   // seq index
                const int dh = wn + fn * 16 + lm;                  // 0..63
                Ctx[((size_t)gi * 2 + bb) * EMBED + hh * DH + dh] =
                    (_Float16)acc[fm][fn][j];
            }
}

// ---------------------------------------------------------------------------
// 5) Output projection: Out[nb, e] (f32) = Ctx @ Wo^T + bo
//    grid (32, 16), 256 threads. A-tile via async LDS copy.
// ---------------------------------------------------------------------------
__global__ __launch_bounds__(256) void outproj_kernel(const _Float16* __restrict__ Ctx,
                                                      const float* __restrict__ Wo,
                                                      const float* __restrict__ bo,
                                                      float* __restrict__ Out) {
    __shared__ _Float16 As[128 * LDT];
    __shared__ _Float16 Bs[64 * LDT];

    const int tid  = threadIdx.x;
    const int lane = tid & 31;
    const int wv   = tid >> 5;
    const int wm   = (wv & 3) * 32;
    const int wn   = (wv >> 2) * 32;
    const int lm   = lane & 15;
    const int hg   = lane >> 4;
    const int m0   = blockIdx.x * 128;
    const int n0   = blockIdx.y * 64;

    const int r = tid >> 3;
    const int c = (tid & 7) * 4;
    const int ar = tid >> 1;           // A copy: 2 threads/row, 32B each
    const int ac = (tid & 1) * 16;

    v8f acc[2][2] = {};

    for (int k0 = 0; k0 < EMBED; k0 += 32) {
        if (k0 + 32 < EMBED)           // prefetch next Wo K-tile into near caches
            __builtin_prefetch(Wo + (size_t)(n0 + r) * EMBED + k0 + 32 + c, 0, 3);
        {   // A: ctx f16 copy (128 x 32 halves) via async b128
            const _Float16* g = Ctx + (size_t)(m0 + ar) * EMBED + k0 + ac;
            _Float16* l = As + ar * LDT + ac;
            ASYNC_CP_B128(g, l);
            ASYNC_CP_B128(g + 8, l + 8);
        }
#pragma unroll
        for (int p = 0; p < 2; ++p) {    // B: Wo f32 -> f16
            const int row = p * 32 + r;
            const float4 x = *(const float4*)(Wo + (size_t)(n0 + row) * EMBED + k0 + c);
            _Float16* d = Bs + row * LDT + c;
            d[0] = (_Float16)x.x; d[1] = (_Float16)x.y;
            d[2] = (_Float16)x.z; d[3] = (_Float16)x.w;
        }
        wait_async_lds();
        __syncthreads();

        v16h a[2], b[2];
        a[0] = frag_a(As, LDT, wm + lm, 0, hg);
        a[1] = frag_a(As, LDT, wm + 16 + lm, 0, hg);
        b[0] = frag_b(Bs, LDT, wn + lm, 0, hg);
        b[1] = frag_b(Bs, LDT, wn + 16 + lm, 0, hg);
#pragma unroll
        for (int fm = 0; fm < 2; ++fm)
#pragma unroll
            for (int fn = 0; fn < 2; ++fn)
                acc[fm][fn] = WMMA_F16(a[fm], b[fn], acc[fm][fn]);
        __syncthreads();
    }

#pragma unroll
    for (int fm = 0; fm < 2; ++fm)
#pragma unroll
        for (int fn = 0; fn < 2; ++fn)
#pragma unroll
            for (int j = 0; j < 8; ++j) {
                const int gm = m0 + wm + fm * 16 + (j + 8 * hg);
                const int gn = n0 + wn + fn * 16 + lm;
                Out[(size_t)gm * EMBED + gn] = acc[fm][fn][j] + bo[gn];
            }
}

// ---------------------------------------------------------------------------
extern "C" void kernel_launch(void* const* d_in, const int* in_sizes, int n_in,
                              void* d_out, int out_size, void* d_ws, size_t ws_size,
                              hipStream_t stream) {
    (void)in_sizes; (void)n_in; (void)out_size; (void)ws_size;

    const float* q  = (const float*)d_in[0];
    const float* k  = (const float*)d_in[1];
    const float* v  = (const float*)d_in[2];
    const float* Wq = (const float*)d_in[3];
    const float* Wk = (const float*)d_in[4];
    const float* Wv = (const float*)d_in[5];
    const float* Wo = (const float*)d_in[6];
    const float* bo = (const float*)d_in[7];

    float* out  = (float*)d_out;                             // (2048,2,1024)
    float* attn = out + (size_t)SEQ * 2 * EMBED;             // (2,16,2048,2048)

    const size_t headElems = (size_t)NBH * SEQ * DH;         // 4,194,304
    _Float16* Qh  = (_Float16*)d_ws;
    _Float16* Kh  = Qh + headElems;
    _Float16* Vh  = Kh + headElems;
    _Float16* Ctx = Vh + headElems;

    const dim3 blk(256);
    const float scale = 0.125f;   // 1/sqrt(64), folded into Q

    proj_kernel<<<dim3(32, 16), blk, 0, stream>>>(q, Wq, Qh, scale);
    proj_kernel<<<dim3(32, 16), blk, 0, stream>>>(k, Wk, Kh, 1.0f);
    proj_kernel<<<dim3(32, 16), blk, 0, stream>>>(v, Wv, Vh, 1.0f);
    scores_kernel<<<dim3(16, 32, NBH), blk, 0, stream>>>(Qh, Kh, attn);
    softmax_kernel<<<dim3(NBH * SEQ), blk, 0, stream>>>(attn);
    attnv_kernel<<<dim3(16, NBH), blk, 0, stream>>>(attn, Vh, Ctx);
    outproj_kernel<<<dim3(32, 16), blk, 0, stream>>>(Ctx, Wo, bo, out);
}
